// VisionTransformer_28527172780144
// MI455X (gfx1250) — matrix operand order repaired and checked by reading, not compiled, not run on hardware
//
#include <hip/hip_runtime.h>
#include <hip/hip_bf16.h>

#define HIDD   768
#define HEADS  12
#define SEQ    197
#define SEQP   256
#define NB     8
#define PATCHD 256
#define NPAT   196
#define OUTD   1000
#define HH     (HEADS * HIDD)   // 9216
#define NBLK   6

typedef unsigned short u16;
typedef __bf16 v16bf __attribute__((ext_vector_type(16)));
typedef float  v8f   __attribute__((ext_vector_type(8)));
typedef int    ivec4 __attribute__((ext_vector_type(4)));

union frag16 { ivec4 q[2]; v16bf v; };

__device__ __forceinline__ u16 f2bf(float f) {
  unsigned u = __float_as_uint(f);
  u += 0x7fffu + ((u >> 16) & 1u);          // round-to-nearest-even
  return (u16)(u >> 16);
}

// Low 32 bits of a generic pointer to LDS = LDS byte offset (aperture in high bits).
__device__ __forceinline__ unsigned lds_off(const void* p) {
  return (unsigned)(unsigned long long)p;
}

// Async DMA: 16 bytes per lane, global -> LDS. Tracked by ASYNCcnt.
__device__ __forceinline__ void async_cp16(unsigned lds, const u16* g) {
  asm volatile("global_load_async_to_lds_b128 %0, %1, off"
               :: "v"(lds), "v"(g) : "memory");
}
#define WAIT_ASYNC_LE3() asm volatile("s_wait_asynccnt 0x3" ::: "memory")
#define WAIT_ASYNC_0()   asm volatile("s_wait_asynccnt 0x0" ::: "memory")

__device__ __forceinline__ v8f wmma_bf16(const frag16& a, const frag16& b, v8f c) {
  return __builtin_amdgcn_wmma_f32_16x16x32_bf16(false, a.v, false, b.v,
                                                 (short)0, c, false, false);
}

// ---------------------------------------------------------------------------
// Batched WMMA GEMM:  C[z][M,N] = A[(z/divA)][M,K] * W[(z%modW)][N,K]^T + bias
// bf16 operands, f32 accumulate (v_wmma_f32_16x16x32_bf16).
// Block = 256 thr (8 waves), block tile 128x64, wave tile 32x32 (4 WMMAs/step).
// K-slabs (128x32 of A, 64x32 of W) staged in LDS, double-buffered async DMA.
// Requires M%128==0, N%64==0, K%32==0.
// ---------------------------------------------------------------------------
__global__ __launch_bounds__(256)
void gemm_bf16_wmma(const u16* __restrict__ A, long long sA, int divA,
                    const u16* __restrict__ W, long long sW, int modW,
                    const float* __restrict__ bias, long long sBias, int biasMod,
                    float* __restrict__ C, long long sC,
                    int M, int N, int K)
{
  __shared__ __attribute__((aligned(16))) u16 smA[2][128 * 32];  // 8 KB each
  __shared__ __attribute__((aligned(16))) u16 smW[2][64 * 32];   // 4 KB each

  const int z = blockIdx.z;
  A += (long long)(z / divA) * sA;
  W += (long long)(z % modW) * sW;
  C += (long long)z * sC;
  const float* bp = bias ? (bias + (long long)(z % biasMod) * sBias) : nullptr;

  const int tid  = threadIdx.x;
  const int lane = tid & 31;
  const int wave = tid >> 5;
  const int rowB = blockIdx.y * 128;
  const int colB = blockIdx.x * 64;
  const int wm = wave >> 1;          // 0..3 -> 32-row slice
  const int wn = wave & 1;           // 0..1 -> 32-col slice

  // DMA mapping: thread t -> tile row t>>2 (0..63), 16B chunk (t&3)*8 elems.
  // A slab is 128 rows: thread also covers row (t>>2)+64.
  const int cpRow = tid >> 2;
  const int cpCh  = (tid & 3) * 8;
  const u16* gA0 = A + (long long)(rowB + cpRow) * K + cpCh;
  const u16* gA1 = A + (long long)(rowB + cpRow + 64) * K + cpCh;
  const u16* gW  = W + (long long)(colB + cpRow) * K + cpCh;
  const int eA0 = cpRow * 32 + cpCh;
  const int eA1 = (cpRow + 64) * 32 + cpCh;
  const int eW  = cpRow * 32 + cpCh;

  // Fragment addressing (ISA 16-bit WMMA layouts):
  //  A: lane -> M = lane&15; K chunks {0..7,16..23} (lanes<16) / {8..15,24..31}
  //  B: lane -> N = lane&15; K chunks {0..7,16..23} (lanes<16) / {8..15,24..31}
  const int am0 = wm * 32 + (lane & 15);
  const int am1 = am0 + 16;
  const int ka  = (lane >> 4) * 8;
  const int bn0 = wn * 32 + (lane & 15);
  const int bn1 = bn0 + 16;
  const int kb  = (lane >> 4) * 8;

  const int nsteps = K / 32;

  // prologue: DMA slab 0 into buffer 0
  async_cp16(lds_off(&smA[0][eA0]), gA0);
  async_cp16(lds_off(&smA[0][eA1]), gA1);
  async_cp16(lds_off(&smW[0][eW]),  gW);

  v8f acc00 = {}; v8f acc01 = {}; v8f acc10 = {}; v8f acc11 = {};

  for (int i = 0; i < nsteps; ++i) {
    const int cur = i & 1;
    if (i + 1 < nsteps) {
      const int nxt = cur ^ 1;
      const long long ko = (long long)(i + 1) * 32;
      async_cp16(lds_off(&smA[nxt][eA0]), gA0 + ko);
      async_cp16(lds_off(&smA[nxt][eA1]), gA1 + ko);
      async_cp16(lds_off(&smW[nxt][eW]),  gW + ko);
      WAIT_ASYNC_LE3();                // slab i retired; slab i+1 in flight
    } else {
      WAIT_ASYNC_0();                  // drain
    }
    __syncthreads();                   // publish slab i to all waves

    const u16* aT = &smA[cur][0];
    const u16* wT = &smW[cur][0];
    frag16 a0, a1, b0, b1;
    a0.q[0] = *(const ivec4*)(aT + am0 * 32 + ka);
    a0.q[1] = *(const ivec4*)(aT + am0 * 32 + ka + 16);
    a1.q[0] = *(const ivec4*)(aT + am1 * 32 + ka);
    a1.q[1] = *(const ivec4*)(aT + am1 * 32 + ka + 16);
    b0.q[0] = *(const ivec4*)(wT + bn0 * 32 + kb);
    b0.q[1] = *(const ivec4*)(wT + bn0 * 32 + kb + 16);
    b1.q[0] = *(const ivec4*)(wT + bn1 * 32 + kb);
    b1.q[1] = *(const ivec4*)(wT + bn1 * 32 + kb + 16);

    acc00 = wmma_bf16(a0, b0, acc00);
    acc01 = wmma_bf16(a0, b1, acc01);
    acc10 = wmma_bf16(a1, b0, acc10);
    acc11 = wmma_bf16(a1, b1, acc11);
    __syncthreads();                   // all waves done reading before rewrite
  }

  // C/D layout: VGPR r -> M = base + r (+8 for lanes 16..31), N = base + (lane&15)
  const int crow  = rowB + wm * 32 + ((lane >> 4) << 3);
  const int ccol0 = colB + wn * 32 + (lane & 15);
  const int ccol1 = ccol0 + 16;
  const float bv0 = bp ? bp[ccol0] : 0.0f;
  const float bv1 = bp ? bp[ccol1] : 0.0f;
#pragma unroll
  for (int r = 0; r < 8; ++r) {
    const long long r0 = (long long)(crow + r) * N;
    const long long r1 = (long long)(crow + 16 + r) * N;
    C[r0 + ccol0] = acc00[r] + bv0;
    C[r0 + ccol1] = acc01[r] + bv1;
    C[r1 + ccol0] = acc10[r] + bv0;
    C[r1 + ccol1] = acc11[r] + bv1;
  }
}

// ---------------------------------------------------------------------------
// Elementwise f32 -> bf16
// ---------------------------------------------------------------------------
__global__ void f32_to_bf16(const float* __restrict__ in, u16* __restrict__ out,
                            long long n)
{
  long long i = (long long)blockIdx.x * blockDim.x + threadIdx.x;
  const long long stride = (long long)gridDim.x * blockDim.x;
  for (; i < n; i += stride) out[i] = f2bf(in[i]);
}

// Patches: X[8,224,224] flat-reshaped to [8,196,256]; pad rows to 256, bf16.
__global__ void patch_to_bf16(const float* __restrict__ X, u16* __restrict__ out)
{
  long long i = (long long)blockIdx.x * blockDim.x + threadIdx.x;
  if (i >= (long long)NB * SEQP * PATCHD) return;
  const int n = (int)(i / (SEQP * PATCHD));
  const int r = (int)(i % (SEQP * PATCHD));
  const int row = r / PATCHD, c = r % PATCHD;
  out[i] = (row < NPAT) ? f2bf(X[(long long)n * NPAT * PATCHD + (long long)row * PATCHD + c]) : (u16)0;
}

// Z[n,0,:]=cls+pos0 ; Z[n,1+p,:]=E[n,p,:]+pos[1+p] ; pad rows = 0
__global__ void build_Z(const float* __restrict__ E, const float* __restrict__ cls,
                        const float* __restrict__ pos, float* __restrict__ Z)
{
  long long i = (long long)blockIdx.x * blockDim.x + threadIdx.x;
  if (i >= (long long)NB * SEQP * HIDD) return;
  const int n = (int)(i / (SEQP * HIDD));
  const int r = (int)(i % (SEQP * HIDD));
  const int s = r / HIDD, d = r % HIDD;
  float v = 0.0f;
  if (s == 0)            v = cls[d] + pos[d];
  else if (s < SEQ)      v = E[(long long)n * SEQP * HIDD + (long long)(s - 1) * HIDD + d] + pos[s * HIDD + d];
  Z[i] = v;
}

// LayerNorm over (SEQ,HID) per batch: out_bf16 = ((x(+res))-mean)*rstd*g + b
__global__ __launch_bounds__(256)
void ln2d_to_bf16(const float* __restrict__ x, const float* __restrict__ res,
                  const float* __restrict__ g, const float* __restrict__ b,
                  u16* __restrict__ out)
{
  const int n = blockIdx.x;
  const int tid = threadIdx.x;
  const float* xb = x + (long long)n * SEQP * HIDD;
  const float* rb = res ? res + (long long)n * SEQP * HIDD : nullptr;
  u16* ob = out + (long long)n * SEQP * HIDD;

  __shared__ float s1[256], s2[256];
  float sum = 0.0f, sq = 0.0f;
  for (int i = tid; i < SEQ * HIDD; i += 256) {
    float v = xb[i] + (rb ? rb[i] : 0.0f);
    sum += v; sq += v * v;
  }
  s1[tid] = sum; s2[tid] = sq;
  __syncthreads();
  for (int off = 128; off > 0; off >>= 1) {
    if (tid < off) { s1[tid] += s1[tid + off]; s2[tid] += s2[tid + off]; }
    __syncthreads();
  }
  const float inv = 1.0f / (float)(SEQ * HIDD);
  const float m = s1[0] * inv;
  const float var = s2[0] * inv - m * m;
  const float rstd = rsqrtf(var + 1e-5f);

  for (int i = tid; i < SEQ * HIDD; i += 256) {
    float v = xb[i] + (rb ? rb[i] : 0.0f);
    ob[i] = f2bf((v - m) * rstd * g[i] + b[i]);
  }
  for (int i = SEQ * HIDD + tid; i < SEQP * HIDD; i += 256) ob[i] = 0;
}

// Row softmax over t<197 with scale 1/HID^2; pads (t>=197, rows>=197) -> 0.
__global__ __launch_bounds__(256)
void softmax_to_bf16(const float* __restrict__ S, u16* __restrict__ A)
{
  const int row = blockIdx.x;
  const int z = blockIdx.y;
  const int t = threadIdx.x;
  u16* arow = A + (long long)z * SEQP * SEQP + (long long)row * SEQP;
  if (row >= SEQ) { arow[t] = 0; return; }
  const float* srow = S + (long long)z * SEQP * SEQP + (long long)row * SEQP;
  const float scale = 1.0f / ((float)HIDD * (float)HIDD);

  __shared__ float red[256];
  const bool valid = (t < SEQ);
  float v = valid ? srow[t] * scale : -3.0e38f;
  red[t] = v; __syncthreads();
  for (int off = 128; off > 0; off >>= 1) {
    if (t < off) red[t] = fmaxf(red[t], red[t + off]);
    __syncthreads();
  }
  const float mx = red[0];
  __syncthreads();
  float e = valid ? __expf(v - mx) : 0.0f;
  red[t] = e; __syncthreads();
  for (int off = 128; off > 0; off >>= 1) {
    if (t < off) red[t] += red[t + off];
    __syncthreads();
  }
  arow[t] = f2bf(e / red[0]);
}

// V [z][SEQP,HID] f32 -> V^T [z][HID,SEQP] bf16 (weight form for A@V GEMM)
__global__ void transpose_to_bf16(const float* __restrict__ in, u16* __restrict__ out)
{
  const int z = blockIdx.z;
  long long i = (long long)blockIdx.x * blockDim.x + threadIdx.x;
  if (i >= (long long)SEQP * HIDD) return;
  const int s = (int)(i / HIDD), e = (int)(i % HIDD);
  out[(long long)z * HIDD * SEQP + (long long)e * SEQP + s] =
      f2bf(in[(long long)z * SEQP * HIDD + i]);
}

// h [n,h][SEQP,HID] f32 -> hcat [n][SEQP, H*HID] bf16
__global__ void concat_to_bf16(const float* __restrict__ in, u16* __restrict__ out)
{
  const int z = blockIdx.z;            // n*HEADS + h
  const int n = z / HEADS, h = z % HEADS;
  long long i = (long long)blockIdx.x * blockDim.x + threadIdx.x;
  if (i >= (long long)SEQP * HIDD) return;
  const int s = (int)(i / HIDD), e = (int)(i % HIDD);
  out[((long long)n * SEQP + s) * HH + h * HIDD + e] =
      f2bf(in[(long long)z * SEQP * HIDD + i]);
}

// out[n,o] = tanh(Z[n,0,:] . Wh[o,:] + bh[o])
__global__ void head_kernel(const float* __restrict__ Z, const float* __restrict__ Wh,
                            const float* __restrict__ bh, float* __restrict__ out)
{
  const int idx = blockIdx.x * blockDim.x + threadIdx.x;
  if (idx >= NB * OUTD) return;
  const int n = idx / OUTD, o = idx % OUTD;
  const float* zr = Z + (long long)n * SEQP * HIDD;
  const float* wr = Wh + (long long)o * HIDD;
  float acc = bh[o];
  for (int d = 0; d < HIDD; ++d) acc += zr[d] * wr[d];
  out[idx] = tanhf(acc);
}

// ---------------------------------------------------------------------------
extern "C" void kernel_launch(void* const* d_in, const int* in_sizes, int n_in,
                              void* d_out, int out_size, void* d_ws, size_t ws_size,
                              hipStream_t stream)
{
  const float* X    = (const float*)d_in[0];
  const float* Wpat = (const float*)d_in[1];
  const float* bpat = (const float*)d_in[2];
  const float* cls  = (const float*)d_in[3];
  const float* pos  = (const float*)d_in[4];
  const float* ln1g = (const float*)d_in[5];
  const float* ln1b = (const float*)d_in[6];
  const float* Wq   = (const float*)d_in[7];
  const float* bq   = (const float*)d_in[8];
  const float* Wk   = (const float*)d_in[9];
  const float* bk   = (const float*)d_in[10];
  const float* Wv   = (const float*)d_in[11];
  const float* bv   = (const float*)d_in[12];
  const float* Wo   = (const float*)d_in[13];
  const float* bo   = (const float*)d_in[14];
  const float* ln2g = (const float*)d_in[15];
  const float* ln2b = (const float*)d_in[16];
  const float* W2   = (const float*)d_in[17];
  const float* b2   = (const float*)d_in[18];
  const float* Wh   = (const float*)d_in[19];
  const float* bh   = (const float*)d_in[20];
  float* outp = (float*)d_out;

  char* ws = (char*)d_ws;
  size_t off = 0;
  auto carve = [&](size_t bytes) -> void* {
    off = (off + 255) & ~(size_t)255;
    void* p = ws + off;
    off += bytes;
    return p;
  };

  const long long SH  = (long long)SEQP * HIDD;      // 196608
  const long long SS  = (long long)SEQP * SEQP;      // 65536
  const int ZHN = NB * HEADS;                        // 96

  u16* wpat_bf = (u16*)carve((size_t)HIDD * PATCHD * 2);
  u16* wq_bf   = (u16*)carve((size_t)HEADS * HIDD * HIDD * 2);
  u16* wk_bf   = (u16*)carve((size_t)HEADS * HIDD * HIDD * 2);
  u16* wv_bf   = (u16*)carve((size_t)HEADS * HIDD * HIDD * 2);
  u16* wo_bf   = (u16*)carve((size_t)HIDD * HH * 2);
  u16* w2_bf   = (u16*)carve((size_t)HIDD * HIDD * 2);
  u16* apat    = (u16*)carve((size_t)NB * SEQP * PATCHD * 2);
  float* Epad  = (float*)carve((size_t)NB * SH * 4);
  float* Zbuf  = (float*)carve((size_t)NB * SH * 4);
  u16* xln     = (u16*)carve((size_t)NB * SH * 2);
  float* bigf  = (float*)carve((size_t)ZHN * SH * 4);   // q/k/v/h f32 (reused)
  u16* q_bf    = (u16*)carve((size_t)ZHN * SH * 2);     // reused as hcat bf16
  u16* k_bf    = (u16*)carve((size_t)ZHN * SH * 2);
  u16* vT_bf   = (u16*)carve((size_t)ZHN * SH * 2);
  float* scor  = (float*)carve((size_t)ZHN * SS * 4);
  u16* A_bf    = (u16*)carve((size_t)ZHN * SS * 2);
  float* attn  = (float*)carve((size_t)NB * SH * 4);
  u16* y_bf    = (u16*)carve((size_t)NB * SH * 2);
  u16* hcat_bf = q_bf;
  (void)ws_size; (void)in_sizes; (void)n_in; (void)out_size;

  auto conv = [&](const float* src, u16* dst, long long n) {
    int blocks = (int)((n + 255) / 256);
    if (blocks > 4096) blocks = 4096;
    f32_to_bf16<<<blocks, 256, 0, stream>>>(src, dst, n);
  };
  auto gemm = [&](const u16* A, long long sA, int divA,
                  const u16* W, long long sW, int modW,
                  const float* bias, long long sBias, int biasMod,
                  float* C, long long sC, int M, int N, int K, int Zc) {
    dim3 g(N / 64, M / 128, Zc);
    gemm_bf16_wmma<<<g, 256, 0, stream>>>(A, sA, divA, W, sW, modW,
                                          bias, sBias, biasMod, C, sC, M, N, K);
  };

  // --- weight conversion (stateless: every call) ---
  conv(Wpat, wpat_bf, (long long)HIDD * PATCHD);
  conv(Wq,   wq_bf,   (long long)HEADS * HIDD * HIDD);
  conv(Wk,   wk_bf,   (long long)HEADS * HIDD * HIDD);
  conv(Wv,   wv_bf,   (long long)HEADS * HIDD * HIDD);
  conv(Wo,   wo_bf,   (long long)HIDD * HH);
  conv(W2,   w2_bf,   (long long)HIDD * HIDD);

  // --- patch embedding + Z assembly ---
  patch_to_bf16<<<(NB * SEQP * PATCHD) / 256, 256, 0, stream>>>(X, apat);
  gemm(apat, (long long)SEQP * PATCHD, 1, wpat_bf, 0, 1,
       bpat, 0, 1, Epad, SH, SEQP, HIDD, PATCHD, NB);
  build_Z<<<(NB * SEQP * HIDD) / 256, 256, 0, stream>>>(Epad, cls, pos, Zbuf);

  // --- transformer blocks (same weights each iteration, as in source) ---
  for (int blk = 0; blk < NBLK; ++blk) {
    ln2d_to_bf16<<<NB, 256, 0, stream>>>(Zbuf, nullptr, ln1g, ln1b, xln);

    // Q
    gemm(xln, SH, HEADS, wq_bf, (long long)HIDD * HIDD, HEADS,
         bq, HIDD, HEADS, bigf, SH, SEQP, HIDD, HIDD, ZHN);
    conv(bigf, q_bf, (long long)ZHN * SH);
    // K
    gemm(xln, SH, HEADS, wk_bf, (long long)HIDD * HIDD, HEADS,
         bk, HIDD, HEADS, bigf, SH, SEQP, HIDD, HIDD, ZHN);
    conv(bigf, k_bf, (long long)ZHN * SH);
    // V (then transpose into weight form [HID, SEQP])
    gemm(xln, SH, HEADS, wv_bf, (long long)HIDD * HIDD, HEADS,
         bv, HIDD, HEADS, bigf, SH, SEQP, HIDD, HIDD, ZHN);
    {
      dim3 g((unsigned)(SH / 256), 1, ZHN);
      transpose_to_bf16<<<g, 256, 0, stream>>>(bigf, vT_bf);
    }

    // scores[s,t] = q[s,:] . k[t,:]
    gemm(q_bf, SH, 1, k_bf, SH, ZHN, nullptr, 0, 1,
         scor, SS, SEQP, SEQP, HIDD, ZHN);
    {
      dim3 g(SEQP, ZHN);
      softmax_to_bf16<<<g, 256, 0, stream>>>(scor, A_bf);
    }
    // h = A @ v  (W form = v^T)
    gemm(A_bf, SS, 1, vT_bf, SH, ZHN, nullptr, 0, 1,
         bigf, SH, SEQP, HIDD, SEQP, ZHN);
    {
      dim3 g((unsigned)(SH / 256), 1, ZHN);
      concat_to_bf16<<<g, 256, 0, stream>>>(bigf, hcat_bf);
    }
    // attn = hcat @ Wo^T + bo
    gemm(hcat_bf, (long long)SEQP * HH, 1, wo_bf, 0, 1,
         bo, 0, 1, attn, SH, SEQP, HIDD, HH, NB);
    // y = LN2(attn + Z) ; Z = y @ W2^T + b2
    ln2d_to_bf16<<<NB, 256, 0, stream>>>(attn, Zbuf, ln2g, ln2b, y_bf);
    gemm(y_bf, SH, 1, w2_bf, 0, 1,
         b2, 0, 1, Zbuf, SH, SEQP, HIDD, HIDD, NB);
  }

  head_kernel<<<(NB * OUTD + 255) / 256, 256, 0, stream>>>(Zbuf, Wh, bh, outp);
}
